// LSTMModel_80229989089420
// MI455X (gfx1250) — matrix operand order repaired and checked by reading, compiled, not run on hardware
//
#include <hip/hip_runtime.h>
#include <hip/hip_bf16.h>

typedef __attribute__((ext_vector_type(16))) _Float16 v16h;
typedef __attribute__((ext_vector_type(8)))  _Float16 v8h;
typedef __attribute__((ext_vector_type(8)))  float    v8f;

#define BB 2048
#define TT 512
#define FF 64
#define HH 30
#define GG 120   // 4*H
#define NL 5
#define TILE_B 16
#define TPB 256  // 8 waves (wave32)
#define A_STRIDE 104   // halves; 208B rows -> 16B aligned, conflict-free frag loads
#define W_STRIDE 96    // halves; 192B rows -> 16B aligned
#define G_STRIDE 132   // floats; breaks the +8-row bank alias on D scatter

// --- fast gate math: single-instruction transcendentals on the serial path ---
#if defined(__has_builtin)
#if __has_builtin(__builtin_amdgcn_tanhf)
#define LSTM_HAS_HW_TANH 1
#endif
#endif

__device__ __forceinline__ float ftanh(float x) {
#ifdef LSTM_HAS_HW_TANH
  return __builtin_amdgcn_tanhf(x);            // v_tanh_f32
#else
  // stable: 1 - 2/(1+e^{2x});  v_exp_f32 + v_rcp_f32 (no IEEE div sequence)
  return 1.f - 2.f * __builtin_amdgcn_rcpf(1.f + __expf(2.f * x));
#endif
}
__device__ __forceinline__ float fsigm(float x) {
#ifdef LSTM_HAS_HW_TANH
  return __builtin_fmaf(__builtin_amdgcn_tanhf(0.5f * x), 0.5f, 0.5f);
#else
  return __builtin_amdgcn_rcpf(1.f + __expf(-x));  // stable for all x
#endif
}

// Build a v16h WMMA fragment from two 16-byte aligned LDS chunks.
__device__ __forceinline__ v16h load_frag16(const _Float16* p0, const _Float16* p1) {
  v8h lo = *(const v8h*)p0;
  v8h hi = *(const v8h*)p1;
  v16h r;
#pragma unroll
  for (int i = 0; i < 8; ++i) { r[i] = lo[i]; r[i + 8] = hi[i]; }
  return r;
}

__global__ __launch_bounds__(TPB) void lstm_fused(
    const float* __restrict__ x,
    const float* __restrict__ W_ih0, const float* __restrict__ W_hh0,
    const float* __restrict__ b_ih0, const float* __restrict__ b_hh0,
    const float* __restrict__ W_ihr, const float* __restrict__ W_hhr,
    const float* __restrict__ b_ihr, const float* __restrict__ b_hhr,
    _Float16* __restrict__ act)  // [B, T, H] f16 inter-layer buffer (in-place per layer)
{
  __shared__ alignas(16) _Float16 sW[128 * W_STRIDE];    // B operand: sW[n][k] = Wcat[k][n], f16
  __shared__ alignas(16) _Float16 sA[TILE_B * A_STRIDE]; // A operand: [x_t ; h] per batch row
  __shared__ alignas(16) float    sG[TILE_B * G_STRIDE]; // gate pre-activations

  const int tid   = threadIdx.x;
  const int lane  = tid & 31;
  const int wave  = tid >> 5;
  const int bBase = blockIdx.x * TILE_B;

  // Elementwise slots: thread owns (b0,j0) and optionally (b1,j1); 16*30 = 480 pairs.
  const int  i0 = tid,        b0 = i0 / HH, j0 = i0 % HH;
  const int  i1 = tid + TPB,  b1 = i1 / HH, j1 = i1 % HH;
  const bool act1 = (i1 < TILE_B * HH);

  // WMMA lane coordinates (ISA 7.12.2 layouts).
  const int M      = lane & 15;
  const int hiHalf = lane >> 4;  // 0 or 1

  for (int l = 0; l < NL; ++l) {
    const int   I     = (l == 0) ? FF : HH;   // input width this layer
    const int   hoff  = I;                    // h occupies A columns [I, I+30)
    const bool  three = (l == 0);             // 3 K-slabs (K=96) vs 2 (K=64)
    const float* Wih = (l == 0) ? W_ih0 : (W_ihr + (size_t)(l - 1) * GG * HH);
    const float* Whh = (l == 0) ? W_hh0 : (W_hhr + (size_t)(l - 1) * GG * HH);
    const float* bih = (l == 0) ? b_ih0 : (b_ihr + (size_t)(l - 1) * GG);
    const float* bhh = (l == 0) ? b_hh0 : (b_hhr + (size_t)(l - 1) * GG);

    __syncthreads();  // previous layer fully done with sW/sA/sG

    // Stage weights: sW[n][k] = (k<I ? W_ih[n][k] : k<I+H ? W_hh[n][k-I] : 0); rows n>=120 zero.
    for (int idx = tid; idx < 128 * W_STRIDE; idx += TPB) {
      int n = idx / W_STRIDE, k = idx % W_STRIDE;
      float w = 0.f;
      if (n < GG) {
        if (k < I)            w = Wih[n * I + k];
        else if (k < I + HH)  w = Whh[n * HH + (k - I)];
      }
      sW[idx] = (_Float16)w;
    }
    // Zero A tile: clears K padding and sets h0 = 0.
    for (int idx = tid; idx < TILE_B * A_STRIDE; idx += TPB) sA[idx] = (_Float16)0.f;

    // Per-thread combined bias (constant over t).
    float bI0 = bih[j0] + bhh[j0],           bF0 = bih[30 + j0] + bhh[30 + j0];
    float bG0 = bih[60 + j0] + bhh[60 + j0], bO0 = bih[90 + j0] + bhh[90 + j0];
    float bI1 = 0, bF1 = 0, bG1 = 0, bO1 = 0;
    if (act1) {
      bI1 = bih[j1] + bhh[j1];           bF1 = bih[30 + j1] + bhh[30 + j1];
      bG1 = bih[60 + j1] + bhh[60 + j1]; bO1 = bih[90 + j1] + bhh[90 + j1];
    }
    float c0 = 0.f, c1 = 0.f;  // cell state lives in registers

    __syncthreads();

    // Hoist B fragments out of the time loop (weights constant over t).
    // B frag: lane column n = 16*wave + (lane&15); K {0..7,16..23} / {8..15,24..31} by lane-half.
    const _Float16* wrow = sW + ((wave << 4) + M) * W_STRIDE + (hiHalf << 3);
    const v16h bf0 = load_frag16(wrow +  0, wrow + 16);
    const v16h bf1 = load_frag16(wrow + 32, wrow + 48);
    const v16h bf2 = load_frag16(wrow + 64, wrow + 80);

    // Software prefetch of step-0 inputs into registers.
    float xr0 = 0.f, xr1 = 0.f, xr2 = 0.f, xr3 = 0.f;
    if (l == 0) {
      int i;
      i = tid;       xr0 = x[((size_t)(bBase + (i >> 6)) * TT + 0) * FF + (i & 63)];
      i = tid + 256; xr1 = x[((size_t)(bBase + (i >> 6)) * TT + 0) * FF + (i & 63)];
      i = tid + 512; xr2 = x[((size_t)(bBase + (i >> 6)) * TT + 0) * FF + (i & 63)];
      i = tid + 768; xr3 = x[((size_t)(bBase + (i >> 6)) * TT + 0) * FF + (i & 63)];
    } else {
      xr0 = (float)act[((size_t)(bBase + b0) * TT + 0) * HH + j0];
      if (act1) xr1 = (float)act[((size_t)(bBase + b1) * TT + 0) * HH + j1];
    }

    for (int t = 0; t < TT; ++t) {
      // Commit prefetched inputs into the A tile (x region, cols [0,I)).
      if (l == 0) {
        int i;
        i = tid;       sA[(i >> 6) * A_STRIDE + (i & 63)] = (_Float16)xr0;
        i = tid + 256; sA[(i >> 6) * A_STRIDE + (i & 63)] = (_Float16)xr1;
        i = tid + 512; sA[(i >> 6) * A_STRIDE + (i & 63)] = (_Float16)xr2;
        i = tid + 768; sA[(i >> 6) * A_STRIDE + (i & 63)] = (_Float16)xr3;
      } else {
        sA[b0 * A_STRIDE + j0] = (_Float16)xr0;
        if (act1) sA[b1 * A_STRIDE + j1] = (_Float16)xr1;
      }
      // Issue prefetch for t+1 now; completes under WMMA + gate math.
      if (t + 1 < TT) {
        if (l == 0) {
          int i;
          i = tid;       xr0 = x[((size_t)(bBase + (i >> 6)) * TT + (t + 1)) * FF + (i & 63)];
          i = tid + 256; xr1 = x[((size_t)(bBase + (i >> 6)) * TT + (t + 1)) * FF + (i & 63)];
          i = tid + 512; xr2 = x[((size_t)(bBase + (i >> 6)) * TT + (t + 1)) * FF + (i & 63)];
          i = tid + 768; xr3 = x[((size_t)(bBase + (i >> 6)) * TT + (t + 1)) * FF + (i & 63)];
        } else {
          xr0 = (float)act[((size_t)(bBase + b0) * TT + (t + 1)) * HH + j0];
          if (act1) xr1 = (float)act[((size_t)(bBase + b1) * TT + (t + 1)) * HH + j1];
        }
      }
      __syncthreads();

      // g[16 x 120] = [x_t ; h] x [W_ih ; W_hh]^T, per-wave 16-column slice, K-slab accumulate.
      // A frag: row M, K-halves {0..7,16..23} / {8..15,24..31} by lane-half.
      // Issue all fragment loads up front so ds latency overlaps, then chain the WMMAs.
      const _Float16* arow = sA + M * A_STRIDE + (hiHalf << 3);
      v8f acc = {};
      if (three) {
        v16h a0 = load_frag16(arow +  0, arow + 16);
        v16h a1 = load_frag16(arow + 32, arow + 48);
        v16h a2 = load_frag16(arow + 64, arow + 80);
        acc = __builtin_amdgcn_wmma_f32_16x16x32_f16(false, a0, false, bf0, (short)0, acc, false, false);
        acc = __builtin_amdgcn_wmma_f32_16x16x32_f16(false, a1, false, bf1, (short)0, acc, false, false);
        acc = __builtin_amdgcn_wmma_f32_16x16x32_f16(false, a2, false, bf2, (short)0, acc, false, false);
      } else {
        v16h a0 = load_frag16(arow +  0, arow + 16);
        v16h a1 = load_frag16(arow + 32, arow + 48);
        acc = __builtin_amdgcn_wmma_f32_16x16x32_f16(false, a0, false, bf0, (short)0, acc, false, false);
        acc = __builtin_amdgcn_wmma_f32_16x16x32_f16(false, a1, false, bf1, (short)0, acc, false, false);
      }
      // Scatter D: element v is (M = v + 8*hiHalf, N = lane&15).
      const int gn = (wave << 4) + M;
#pragma unroll
      for (int v = 0; v < 8; ++v)
        sG[(v + (hiHalf << 3)) * G_STRIDE + gn] = acc[v];

      __syncthreads();

      // Gate nonlinearities + state update; h goes into A's h-region and the f16 activation buffer.
      {
        const float* g = sG + b0 * G_STRIDE + j0;
        float ig = fsigm(g[0]  + bI0);
        float fg = fsigm(g[30] + bF0);
        float gg = ftanh(g[60] + bG0);
        float og = fsigm(g[90] + bO0);
        c0 = fg * c0 + ig * gg;
        float h = og * ftanh(c0);
        sA[b0 * A_STRIDE + hoff + j0] = (_Float16)h;
        act[((size_t)(bBase + b0) * TT + t) * HH + j0] = (_Float16)h;
      }
      if (act1) {
        const float* g = sG + b1 * G_STRIDE + j1;
        float ig = fsigm(g[0]  + bI1);
        float fg = fsigm(g[30] + bF1);
        float gg = ftanh(g[60] + bG1);
        float og = fsigm(g[90] + bO1);
        c1 = fg * c1 + ig * gg;
        float h = og * ftanh(c1);
        sA[b1 * A_STRIDE + hoff + j1] = (_Float16)h;
        act[((size_t)(bBase + b1) * TT + t) * HH + j1] = (_Float16)h;
      }
    }  // t
  }  // layer
}

// out[b, c] = relu(h_last[b,:]) . W_fc[c,:] + b_fc[c]
__global__ __launch_bounds__(256) void fc_kernel(const _Float16* __restrict__ act,
                                                 const float* __restrict__ W_fc,
                                                 const float* __restrict__ b_fc,
                                                 float* __restrict__ out) {
  int b = blockIdx.x * blockDim.x + threadIdx.x;
  if (b >= BB) return;
  const _Float16* hp = act + ((size_t)b * TT + (TT - 1)) * HH;
  float h[HH];
#pragma unroll
  for (int j = 0; j < HH; ++j) {
    float v = (float)hp[j];
    h[j] = v > 0.f ? v : 0.f;
  }
#pragma unroll
  for (int c = 0; c < 2; ++c) {
    float s = b_fc[c];
#pragma unroll
    for (int j = 0; j < HH; ++j) s += h[j] * W_fc[c * HH + j];
    out[b * 2 + c] = s;
  }
}

extern "C" void kernel_launch(void* const* d_in, const int* in_sizes, int n_in,
                              void* d_out, int out_size, void* d_ws, size_t ws_size,
                              hipStream_t stream) {
  (void)in_sizes; (void)n_in; (void)out_size; (void)ws_size;
  const float* x     = (const float*)d_in[0];
  const float* W_ih0 = (const float*)d_in[1];
  const float* W_hh0 = (const float*)d_in[2];
  const float* b_ih0 = (const float*)d_in[3];
  const float* b_hh0 = (const float*)d_in[4];
  const float* W_ihr = (const float*)d_in[5];
  const float* W_hhr = (const float*)d_in[6];
  const float* b_ihr = (const float*)d_in[7];
  const float* b_hhr = (const float*)d_in[8];
  const float* W_fc  = (const float*)d_in[9];
  const float* b_fc  = (const float*)d_in[10];
  float*    out = (float*)d_out;
  _Float16* act = (_Float16*)d_ws;  // needs B*T*H*2 = 60 MB of workspace

  lstm_fused<<<BB / TILE_B, TPB, 0, stream>>>(x, W_ih0, W_hh0, b_ih0, b_hh0,
                                              W_ihr, W_hhr, b_ihr, b_hhr, act);
  fc_kernel<<<(BB + 255) / 256, 256, 0, stream>>>(act, W_fc, b_fc, out);
}
